// Stage2GCNEncoder_43344809951810
// MI455X (gfx1250) — matrix operand order repaired and checked by reading, not compiled, not run on hardware
//
#include <hip/hip_runtime.h>
#include <math.h>

#define N_NODES     25000
#define N_NODES_PAD 25008      // padded to multiple of 16 for tile stores
#define N_EDGES     400000     // divisible by 16 -> edge tiles need no guards
#define N_GRAPHS    64
#define IN_F        128
#define HID         64
#define OUT_F       128
#define HEADS       8
#define HC          512        // HEADS*HID
#define TILE_ELEMS  (16 * IN_F)   // one 16-row f16 A tile = 2048 elems = 4KB

typedef _Float16 half4  __attribute__((ext_vector_type(4)));
typedef _Float16 half8  __attribute__((ext_vector_type(8)));
typedef _Float16 half16 __attribute__((ext_vector_type(16)));
typedef float    f32x8  __attribute__((ext_vector_type(8)));
typedef unsigned int u32x4 __attribute__((ext_vector_type(4)));
typedef unsigned int u32x8 __attribute__((ext_vector_type(8)));

union H16 { half16 v; half8 h8[2]; };

__device__ inline void atomicMaxF(float* addr, float val) {
    unsigned int* ua = (unsigned int*)addr;
    unsigned int old = *ua;
    while (__uint_as_float(old) < val) {
        unsigned int assumed = old;
        old = atomicCAS(ua, assumed, __float_as_uint(val));
        if (old == assumed) break;
    }
}

// ---------------------------------------------------------------------------
// K0a: transpose + convert weights f32 [K=128, N=512] -> f16 [N=512, K=128]
// so a WMMA B-fragment (lane: col=lane%16, two runs of 8 contiguous K halves) is two
// contiguous 16-byte global loads.
// ---------------------------------------------------------------------------
__global__ void prep_weights(const float* __restrict__ Wl, const float* __restrict__ Wr,
                             const float* __restrict__ We,
                             _Float16* __restrict__ Wl16, _Float16* __restrict__ Wr16,
                             _Float16* __restrict__ We16) {
    int i = blockIdx.x * blockDim.x + threadIdx.x;     // 0 .. 3*512*128
    if (i >= 3 * HC * IN_F) return;
    int which = i / (HC * IN_F);
    int r = i % (HC * IN_F);
    int n = r / IN_F, k = r % IN_F;
    const float* W = (which == 0) ? Wl : (which == 1) ? Wr : We;
    _Float16*   Wt = (which == 0) ? Wl16 : (which == 1) ? Wr16 : We16;
    Wt[n * IN_F + k] = (_Float16)W[k * HC + n];
}

// K0b: edge_features f32 -> f16 rows (so TDM can DMA raw tiles into LDS).
__global__ void prep_ef16(const float* __restrict__ ef, _Float16* __restrict__ out) {
    long long i = (long long)blockIdx.x * blockDim.x + threadIdx.x;    // float4 slots
    if (i >= (long long)N_EDGES * IN_F / 4) return;
    float4 v = ((const float4*)ef)[i];
    half4 h = { (_Float16)v.x, (_Float16)v.y, (_Float16)v.z, (_Float16)v.w };
    *(half4*)((_Float16*)out + i * 4) = h;
}

__global__ void fill_f32(float* __restrict__ p, float v, int n) {
    int i = blockIdx.x * blockDim.x + threadIdx.x;
    if (i < n) p[i] = v;
}

// ---------------------------------------------------------------------------
// K1: xl = x@Wl + bl, xr = x@Wr + br.
// One wave (32 threads) per 16x16 output tile; K=128 via 4x wmma_f32_16x16x32_f16.
// A tile staged f32->f16 in LDS; fragments read per the 16-bit A layout:
//   lane L holds row M=L%16, K = kc*32 + (L<16 ? {0..7,16..23} : {8..15,24..31})
// ---------------------------------------------------------------------------
__global__ void __launch_bounds__(32)
node_transform(const float* __restrict__ x,
               const _Float16* __restrict__ Wt_l, const _Float16* __restrict__ Wt_r,
               const float* __restrict__ bl, const float* __restrict__ br,
               float* __restrict__ xl, float* __restrict__ xr) {
    __shared__ __align__(32) _Float16 ldsA[TILE_ELEMS];
    const int lane  = threadIdx.x;
    const int row0  = blockIdx.x * 16;
    const int col0  = blockIdx.y * 16;
    const bool right = (blockIdx.z != 0);
    const _Float16* Wt  = right ? Wt_r : Wt_l;
    const float*   bias = right ? br   : bl;
    float*         outp = right ? xr   : xl;

    for (int i = lane; i < TILE_ELEMS; i += 32) {
        int r = i >> 7, k = i & 127;
        int gr = row0 + r;
        float v = (gr < N_NODES) ? x[gr * IN_F + k] : 0.0f;
        ldsA[i] = (_Float16)v;
    }
    __syncthreads();

    const int  mrow = lane & 15;
    const bool hi   = lane >= 16;
    f32x8 acc = {};
#pragma unroll
    for (int kc = 0; kc < 4; ++kc) {
        H16 a;
        const _Float16* Ar = ldsA + mrow * IN_F + kc * 32 + (hi ? 8 : 0);
        a.h8[0] = *(const half8*)(Ar);
        a.h8[1] = *(const half8*)(Ar + 16);
        H16 b;
        const _Float16* Br = Wt + (col0 + mrow) * IN_F + kc * 32 + (hi ? 8 : 0);
        b.h8[0] = *(const half8*)(Br);
        b.h8[1] = *(const half8*)(Br + 16);
        acc = __builtin_amdgcn_wmma_f32_16x16x32_f16(false, a.v, false, b.v,
                                                     (short)0, acc, false, false);
    }
    float bv = bias[col0 + mrow];
#pragma unroll
    for (int r = 0; r < 8; ++r) {
        int m = r + (hi ? 8 : 0);
        outp[(row0 + m) * HC + col0 + mrow] = acc[r] + bv;   // padded buffer: safe
    }
}

// ---------------------------------------------------------------------------
// K2: fused per-edge GEMM + attention logits.
// Block = 256 threads = 8 waves per 16-edge tile; wave h owns head h
// (columns [h*64, h*64+64) = 4 N-tiles), so the att-dot reduces fully in-wave
// via shfl_xor. The 4KB f16 A tile is DMA'd into LDS by the Tensor Data Mover
// (wave 0 issues a 1-D D#: count=1, data_size=2B, tile_dim0=2048, then waits
// on TENSORcnt). ef is never materialized: it lives in WMMA accumulators.
// ---------------------------------------------------------------------------
__global__ void __launch_bounds__(256)
edge_logits(const _Float16* __restrict__ ef16,
            const int* __restrict__ srcArr, const int* __restrict__ dstArr,
            const _Float16* __restrict__ WtE,
            const float* __restrict__ xl, const float* __restrict__ xr,
            const float* __restrict__ att,
            float* __restrict__ ebuf, float* __restrict__ emax) {
    __shared__ __align__(32) _Float16 ldsA[TILE_ELEMS];
    __shared__ int sSrc[16], sDst[16];
    const int tid = threadIdx.x;
    const int e0  = blockIdx.x * 16;

    if (tid < 16) {
        sSrc[tid] = srcArr[e0 + tid];
        sDst[tid] = dstArr[e0 + tid];
    }

    // ---- TDM: DMA the contiguous 4KB tile ef16[e0*128 .. ) into LDS ----
    {
        unsigned lds_off = (unsigned)(size_t)(void*)ldsA;  // generic addr low 32b = LDS offset
        unsigned long long ga = (unsigned long long)(size_t)(ef16 + (size_t)e0 * IN_F);
        u32x4 g0;
        g0[0] = 1u;                                        // count=1 (valid), user mode
        g0[1] = lds_off;                                   // lds_addr
        g0[2] = (unsigned)ga;                              // global_addr[31:0]
        g0[3] = (unsigned)(ga >> 32) | (2u << 30);         // global_addr[56:32] | type=2
        u32x8 g1;
        g1[0] = 1u << 16;                                  // workgroup_mask=0, data_size=2B
        g1[1] = ((unsigned)TILE_ELEMS & 0xFFFFu) << 16;    // tensor_dim0[15:0]
        g1[2] = ((unsigned)TILE_ELEMS >> 16) | (1u << 16); // tensor_dim0[31:16], tensor_dim1=1
        g1[3] = ((unsigned)TILE_ELEMS & 0xFFFFu) << 16;    // tile_dim0 = 2048 elems
        g1[4] = 1u;                                        // tile_dim1=1, tile_dim2=0
        g1[5] = (unsigned)TILE_ELEMS;                      // tensor_dim0_stride[31:0]
        g1[6] = 0u;
        g1[7] = 0u;
        if (tid < 32) {                                    // one wave issues the DMA
            asm volatile("tensor_load_to_lds %0, %1" :: "s"(g0), "s"(g1) : "memory");
            __builtin_amdgcn_s_wait_tensorcnt(0);
        }
    }
    __syncthreads();

    const int  wv   = tid >> 5;          // head this wave owns
    const int  lane = tid & 31;
    const int  mrow = lane & 15;
    const bool hi   = lane >= 16;

    H16 afrag[4];
#pragma unroll
    for (int kc = 0; kc < 4; ++kc) {
        const _Float16* Ar = ldsA + mrow * IN_F + kc * 32 + (hi ? 8 : 0);
        afrag[kc].h8[0] = *(const half8*)(Ar);
        afrag[kc].h8[1] = *(const half8*)(Ar + 16);
    }

    float p[8] = {0, 0, 0, 0, 0, 0, 0, 0};
#pragma unroll
    for (int nt = 0; nt < 4; ++nt) {
        const int col = wv * 64 + nt * 16 + mrow;
        f32x8 acc = {};
#pragma unroll
        for (int kc = 0; kc < 4; ++kc) {
            H16 b;
            const _Float16* Br = WtE + col * IN_F + kc * 32 + (hi ? 8 : 0);
            b.h8[0] = *(const half8*)(Br);
            b.h8[1] = *(const half8*)(Br + 16);
            acc = __builtin_amdgcn_wmma_f32_16x16x32_f16(false, afrag[kc].v, false, b.v,
                                                         (short)0, acc, false, false);
        }
        const float av = att[col];
#pragma unroll
        for (int r = 0; r < 8; ++r) {
            int m = r + (hi ? 8 : 0);
            int s = sSrc[m], d = sDst[m];
            float v = acc[r] + xl[s * HC + col] + xr[d * HC + col];
            v = (v >= 0.0f) ? v : 0.2f * v;              // LeakyReLU(0.2)
            p[r] += v * av;
        }
    }
    // reduce over the 16 lanes of each half (sum over this head's 64 columns)
#pragma unroll
    for (int r = 0; r < 8; ++r) {
        float v = p[r];
        v += __shfl_xor(v, 1, 32);
        v += __shfl_xor(v, 2, 32);
        v += __shfl_xor(v, 4, 32);
        v += __shfl_xor(v, 8, 32);
        p[r] = v;
    }
    if (mrow == 0) {                     // lanes 0 (rows 0-7) and 16 (rows 8-15)
#pragma unroll
        for (int r = 0; r < 8; ++r) {
            int m = r + (hi ? 8 : 0);
            float ev = p[r];
            ebuf[(e0 + m) * HEADS + wv] = ev;
            atomicMaxF(&emax[sDst[m] * HEADS + wv], ev);
        }
    }
}

// K3: ee = exp(e - emax[dst]) in place; esum segment-sum via atomics.
__global__ void softmax_exp(const int* __restrict__ dstArr,
                            float* __restrict__ ebuf,
                            const float* __restrict__ emax,
                            float* __restrict__ esum) {
    int i = blockIdx.x * blockDim.x + threadIdx.x;
    if (i >= N_EDGES * HEADS) return;
    int e = i >> 3, h = i & 7;
    int d = dstArr[e];
    float em = emax[d * HEADS + h];
    if (!(em > -1e29f)) em = 0.0f;                       // finite-clamp
    float v = __expf(ebuf[i] - em);
    ebuf[i] = v;
    atomicAdd(&esum[d * HEADS + h], v);
}

// ---------------------------------------------------------------------------
// K4: head-fused scatter. The reference takes mean over heads right after the
// weighted aggregation, so combine heads in registers per edge:
//   oacc64[dst, c] += sum_h alpha[e,h] * xl[src, h*64 + c]
// -> 8x fewer atomics (25.6M instead of 205M) and a 6.4MB accumulator.
// Block = 16 edges x 16 float4 column-slots; alphas staged once in LDS.
// ---------------------------------------------------------------------------
__global__ void __launch_bounds__(256)
scatter_out(const int* __restrict__ srcArr, const int* __restrict__ dstArr,
            const float* __restrict__ ebuf, const float* __restrict__ esum,
            const float* __restrict__ xl, float* __restrict__ oacc64) {
    __shared__ float sAlpha[16][HEADS];
    __shared__ int sSrc[16], sDst[16];
    const int tid = threadIdx.x;
    const int e0  = blockIdx.x * 16;
    if (tid < 16) {
        sSrc[tid] = srcArr[e0 + tid];
        sDst[tid] = dstArr[e0 + tid];
    }
    __syncthreads();
    if (tid < 16 * HEADS) {
        int m = tid >> 3, h = tid & 7;
        sAlpha[m][h] = ebuf[(e0 + m) * HEADS + h] /
                       (esum[sDst[m] * HEADS + h] + 1e-16f);
    }
    __syncthreads();

    const int m  = tid >> 4;       // edge slot
    const int c4 = tid & 15;       // float4 slot within the 64 head-local cols
    const int s = sSrc[m], d = sDst[m];
    const float4* xl4 = (const float4*)xl;
    float4 accv = {0.0f, 0.0f, 0.0f, 0.0f};
#pragma unroll
    for (int h = 0; h < HEADS; ++h) {
        float a  = sAlpha[m][h];
        float4 v = xl4[s * (HC / 4) + h * 16 + c4];
        accv.x += a * v.x; accv.y += a * v.y;
        accv.z += a * v.z; accv.w += a * v.w;
    }
    float* ob = oacc64 + (size_t)d * HID + c4 * 4;
    atomicAdd(ob + 0, accv.x);
    atomicAdd(ob + 1, accv.y);
    atomicAdd(ob + 2, accv.z);
    atomicAdd(ob + 3, accv.w);
}

// K5: node_emb = tanh(headsum/8 + bias); graph-sum atomics.
__global__ void node_epilogue(const float* __restrict__ oacc64, const float* __restrict__ bias,
                              const int* __restrict__ batch,
                              float* __restrict__ dout, float* __restrict__ gsum,
                              float* __restrict__ cnt) {
    int i = blockIdx.x * blockDim.x + threadIdx.x;
    if (i >= N_NODES * HID) return;
    int n = i >> 6, c = i & 63;
    float t = tanhf(oacc64[n * HID + c] * (1.0f / HEADS) + bias[c]);
    dout[N_GRAPHS * OUT_F + n * HID + c] = t;            // node embeddings
    int b = batch[n];
    atomicAdd(&gsum[b * HID + c], t);
    if (c == 0) atomicAdd(&cnt[b], 1.0f);
}

// K6: graph_emb = tanh((gsum/cnt) @ linW + linb)  (tiny: 1 MF, plain VALU)
__global__ void __launch_bounds__(128)
graph_head(const float* __restrict__ gsum, const float* __restrict__ cnt,
           const float* __restrict__ linW, const float* __restrict__ linb,
           float* __restrict__ dout) {
    __shared__ float gmean[HID];
    int g = blockIdx.x, o = threadIdx.x;
    if (o < HID) {
        float c = cnt[g];
        c = (c > 1.0f) ? c : 1.0f;
        gmean[o] = gsum[g * HID + o] / c;
    }
    __syncthreads();
    float acc = linb[o];
    for (int k = 0; k < HID; ++k) acc += gmean[k] * linW[k * OUT_F + o];
    dout[g * OUT_F + o] = tanhf(acc);
}

extern "C" void kernel_launch(void* const* d_in, const int* in_sizes, int n_in,
                              void* d_out, int out_size, void* d_ws, size_t ws_size,
                              hipStream_t stream) {
    const float* x     = (const float*)d_in[0];
    const int*   ei    = (const int*)  d_in[1];
    const float* ef    = (const float*)d_in[2];
    const int*   batch = (const int*)  d_in[3];
    const float* Wl    = (const float*)d_in[4];
    const float* bl    = (const float*)d_in[5];
    const float* Wr    = (const float*)d_in[6];
    const float* br    = (const float*)d_in[7];
    const float* We    = (const float*)d_in[8];
    const float* att   = (const float*)d_in[9];
    const float* bias  = (const float*)d_in[10];
    const float* linW  = (const float*)d_in[11];
    const float* linb  = (const float*)d_in[12];
    float* out = (float*)d_out;

    const int* src = ei;
    const int* dst = ei + N_EDGES;

    // ---- workspace carve-up (~226 MB) ----
    char* p = (char*)d_ws;
    auto take = [&](size_t bytes) {
        char* r = p;
        p += (bytes + 255) & ~(size_t)255;
        return r;
    };
    _Float16* Wl16 = (_Float16*)take((size_t)HC * IN_F * 2);
    _Float16* Wr16 = (_Float16*)take((size_t)HC * IN_F * 2);
    _Float16* We16 = (_Float16*)take((size_t)HC * IN_F * 2);
    _Float16* ef16 = (_Float16*)take((size_t)N_EDGES * IN_F * 2);
    float* xl     = (float*)take((size_t)N_NODES_PAD * HC * 4);
    float* xr     = (float*)take((size_t)N_NODES_PAD * HC * 4);
    float* ebuf   = (float*)take((size_t)N_EDGES * HEADS * 4);
    float* emax   = (float*)take((size_t)N_NODES * HEADS * 4);
    float* esum   = (float*)take((size_t)N_NODES * HEADS * 4);
    float* oacc64 = (float*)take((size_t)N_NODES * HID * 4);
    float* gsum   = (float*)take((size_t)N_GRAPHS * HID * 4);
    float* cnt    = (float*)take((size_t)N_GRAPHS * 4);

    // ---- init ----
    {
        int n;
        n = N_NODES * HEADS;
        fill_f32<<<(n + 255) / 256, 256, 0, stream>>>(emax, -1e30f, n);
        fill_f32<<<(n + 255) / 256, 256, 0, stream>>>(esum, 0.0f, n);
        n = N_NODES * HID;
        fill_f32<<<(n + 255) / 256, 256, 0, stream>>>(oacc64, 0.0f, n);
        n = N_GRAPHS * HID;
        fill_f32<<<(n + 255) / 256, 256, 0, stream>>>(gsum, 0.0f, n);
        fill_f32<<<1, 256, 0, stream>>>(cnt, 0.0f, N_GRAPHS);
    }

    // ---- pipeline ----
    prep_weights<<<(3 * HC * IN_F + 255) / 256, 256, 0, stream>>>(Wl, Wr, We, Wl16, Wr16, We16);

    long long nef4 = (long long)N_EDGES * IN_F / 4;
    prep_ef16<<<(unsigned)((nef4 + 255) / 256), 256, 0, stream>>>(ef, ef16);

    dim3 g1((N_NODES + 15) / 16, HC / 16, 2);
    node_transform<<<g1, 32, 0, stream>>>(x, Wl16, Wr16, bl, br, xl, xr);

    edge_logits<<<N_EDGES / 16, 256, 0, stream>>>(ef16, src, dst, We16, xl, xr, att, ebuf, emax);

    softmax_exp<<<(N_EDGES * HEADS + 255) / 256, 256, 0, stream>>>(dst, ebuf, emax, esum);

    scatter_out<<<N_EDGES / 16, 256, 0, stream>>>(src, dst, ebuf, esum, xl, oacc64);

    node_epilogue<<<(N_NODES * HID + 255) / 256, 256, 0, stream>>>(oacc64, bias, batch, out, gsum, cnt);

    graph_head<<<N_GRAPHS, 128, 0, stream>>>(gsum, cnt, linW, linb, out);
}
